// TransformerLayer_67637144978464
// MI455X (gfx1250) — matrix operand order repaired and checked by reading, not compiled, not run on hardware
//
#include <hip/hip_runtime.h>
#include <hip/hip_bf16.h>
#include <math.h>
#include <stdint.h>

typedef __attribute__((ext_vector_type(16))) __bf16 v16bf;
typedef __attribute__((ext_vector_type(8)))  __bf16 v8bf;
typedef __attribute__((ext_vector_type(8)))  float  v8f;

constexpr int BATCH = 2;
constexpr int SEQ   = 2048;
constexpr int DM    = 1024;
constexpr int NH    = 16;
constexpr int HD    = 64;
constexpr int TOK   = BATCH * SEQ;           // 4096
constexpr float LNEPS = 1e-5f;
constexpr float NEGBIG = -1e30f;             // finite mask sentinel (no NaN paths)
constexpr size_t OUT0_ELEMS = (size_t)TOK * DM;                 // 4,194,304
constexpr size_t ATT_ELEMS  = (size_t)BATCH * NH * SEQ * SEQ;   // 134,217,728

// ---------------------------------------------------------------- fragments
__device__ inline v16bf pack2(v8bf a, v8bf b) {
  v16bf r;
#pragma unroll
  for (int i = 0; i < 8; ++i) { r[i] = a[i]; r[i + 8] = b[i]; }
  return r;
}
// A-matrix 16x32 bf16 fragment: lane row = lane&15, K chunks [kb,kb+8) and
// [16+kb,16+kb+8) with kb = 8*(lane>>4)   (ISA 7.12.2)
__device__ inline v16bf load_a16(const __bf16* rowp, int hi) {
  v8bf a = *(const v8bf*)(rowp + 8 * hi);
  v8bf b = *(const v8bf*)(rowp + 16 + 8 * hi);
  return pack2(a, b);
}
// B-matrix 32x16 bf16 fragment: lane col = lane&15, K chunks [kb,kb+8) and
// [16+kb,16+kb+8) with kb = 8*(lane>>4)
__device__ inline v16bf load_b16(const __bf16* rowp, int hi) {
  v8bf a = *(const v8bf*)(rowp + 8 * hi);
  v8bf b = *(const v8bf*)(rowp + 16 + 8 * hi);
  return pack2(a, b);
}
__device__ inline v8f wmma_bf16(v16bf a, v16bf b, v8f c) {
  return __builtin_amdgcn_wmma_f32_16x16x32_bf16(false, a, false, b, (short)0, c,
                                                 false, false);
}
// async global->LDS 16B copy (GLOBAL_LOAD_ASYNC_TO_LDS_B128, ASYNCcnt-tracked)
__device__ inline void async_copy_b128(void* lds_ptr, const void* gptr) {
  uint32_t lds_off = (uint32_t)(uintptr_t)lds_ptr;   // low 32b of flat = LDS offset
  asm volatile("global_load_async_to_lds_b128 %0, %1, off"
               :: "v"(lds_off), "v"((uint64_t)(uintptr_t)gptr) : "memory");
}
__device__ inline void wait_async() {
  asm volatile("s_wait_asynccnt 0" ::: "memory");
}

// ---------------------------------------------------------------- utility kernels
__global__ void convert_f32_bf16(const float* __restrict__ in, __bf16* __restrict__ out) {
  size_t i = ((size_t)blockIdx.x * blockDim.x + threadIdx.x) * 4;
  float4 v = *(const float4*)(in + i);
  out[i + 0] = (__bf16)v.x; out[i + 1] = (__bf16)v.y;
  out[i + 2] = (__bf16)v.z; out[i + 3] = (__bf16)v.w;
}

__global__ void zero_f4(float4* __restrict__ p) {
  size_t i = ((size_t)blockIdx.x * blockDim.x + threadIdx.x) * 4;
  float4 z = make_float4(0.f, 0.f, 0.f, 0.f);
  p[i + 0] = z; p[i + 1] = z; p[i + 2] = z; p[i + 3] = z;
}

// wT[n][k] = (bf16) w[k][n], 1024x1024, tiled through LDS for coalescing
__global__ __launch_bounds__(256) void transpose_w(const float* __restrict__ w,
                                                   __bf16* __restrict__ wt) {
  __shared__ float tile[32][33];
  int x = blockIdx.x * 32 + threadIdx.x;   // n
  int y = blockIdx.y * 32 + threadIdx.y;   // k
#pragma unroll
  for (int i = 0; i < 32; i += 8)
    tile[threadIdx.y + i][threadIdx.x] = w[(size_t)(y + i) * DM + x];
  __syncthreads();
  x = blockIdx.y * 32 + threadIdx.x;       // k
  y = blockIdx.x * 32 + threadIdx.y;       // n
#pragma unroll
  for (int i = 0; i < 32; i += 8)
    wt[(size_t)(y + i) * DM + x] = (__bf16)tile[threadIdx.x][threadIdx.y + i];
}

// ---------------------------------------------------------------- GEMM (bf16 WMMA)
// C[M=4096, N=1024] = A @ Wt^T (Wt stored [N][K]).
// 128x128 block tile, 8 waves as 4(M)x2(N): each wave 32x64 = 8 WMMA / k-step.
// Tiles staged with GLOBAL_LOAD_ASYNC_TO_LDS_B128 (ASYNCcnt), double-buffered,
// one barrier per k-step: prefetch of buf^1 races only with reads of buf.
// MODE 0: out bf16 per-head [b][h][s][hd] * oscale   (Q with 1/sqrt(hd), K with 1)
// MODE 1: out bf16 per-head transposed [b][h][hd][s] (V^T)
// MODE 2: outF = acc + resid
// MODE 3: outB = gelu(acc + bias)
// MODE 4: outF = acc + bias + resid
template <int MODE>
__global__ __launch_bounds__(256) void gemm_bf16(
    const __bf16* __restrict__ A, const __bf16* __restrict__ Wt,
    const float* __restrict__ bias, const float* __restrict__ resid,
    float* __restrict__ outF, __bf16* __restrict__ outB, float oscale) {
  __shared__ __attribute__((aligned(16))) __bf16 As[2][128][40];
  __shared__ __attribute__((aligned(16))) __bf16 Bs[2][128][40];
  int tid = threadIdx.x;
  int lane = tid & 31;
  int hi = (lane >> 4) & 1, lo = lane & 15;
  int wave = __builtin_amdgcn_readfirstlane(tid >> 5);
  int wm = wave & 3, wn = wave >> 2;
  int m0 = blockIdx.x * 128, n0 = blockIdx.y * 128;

  auto gload_async = [&](int kk, int buf) {
#pragma unroll
    for (int c = 0; c < 2; ++c) {
      int ch = c * 256 + tid;                // 512 chunks of 8 bf16 per matrix
      int r = ch >> 2, col = (ch & 3) * 8;
      async_copy_b128(&As[buf][r][col], &A[(size_t)(m0 + r) * DM + kk + col]);
      async_copy_b128(&Bs[buf][r][col], &Wt[(size_t)(n0 + r) * DM + kk + col]);
    }
  };

  v8f acc[2][4] = {};
  gload_async(0, 0);
  wait_async();
  __syncthreads();
  constexpr int NIT = DM / 32;
  for (int it = 0; it < NIT; ++it) {
    int buf = it & 1;
    if (it + 1 < NIT) gload_async((it + 1) * 32, buf ^ 1);  // async prefetch
    v16bf af0 = load_a16(&As[buf][wm * 32 + lo][0], hi);
    v16bf af1 = load_a16(&As[buf][wm * 32 + 16 + lo][0], hi);
#pragma unroll
    for (int nt = 0; nt < 4; ++nt) {
      v16bf bf_ = load_b16(&Bs[buf][wn * 64 + nt * 16 + lo][0], hi);
      acc[0][nt] = wmma_bf16(af0, bf_, acc[0][nt]);
      acc[1][nt] = wmma_bf16(af1, bf_, acc[1][nt]);
    }
    if (it + 1 < NIT) {
      wait_async();       // my async writes to buf^1 landed
      __syncthreads();    // everyone's landed; also fences reads of buf
    }
  }

#pragma unroll
  for (int mi = 0; mi < 2; ++mi) {
#pragma unroll
    for (int nt = 0; nt < 4; ++nt) {
#pragma unroll
      for (int j = 0; j < 8; ++j) {
        int m = m0 + wm * 32 + mi * 16 + j + 8 * hi;  // C: M = j + 8*(lane>=16)
        int n = n0 + wn * 64 + nt * 16 + lo;          //    N = lane&15
        float v = acc[mi][nt][j];
        if constexpr (MODE == 0) {
          int b = m >> 11, s = m & (SEQ - 1), h = n >> 6, d = n & 63;
          outB[((size_t)(b * NH + h) * SEQ + s) * HD + d] = (__bf16)(v * oscale);
        } else if constexpr (MODE == 1) {
          int b = m >> 11, s = m & (SEQ - 1), h = n >> 6, d = n & 63;
          outB[((size_t)(b * NH + h) * HD + d) * SEQ + s] = (__bf16)v;
        } else if constexpr (MODE == 2) {
          outF[(size_t)m * DM + n] = v + resid[(size_t)m * DM + n];
        } else if constexpr (MODE == 3) {
          float t = v + bias[n];
          outB[(size_t)m * DM + n] =
              (__bf16)(0.5f * t * (1.0f + erff(t * 0.70710678118f)));
        } else {
          outF[(size_t)m * DM + n] = v + bias[n] + resid[(size_t)m * DM + n];
        }
      }
    }
  }
}

// ---------------------------------------------------------------- attention
// One wave per 16-query tile of one (b,h); (b,h,qt) forced uniform via
// readfirstlane so all loop control is scalar (no EXEC churn around WMMA).
// Branchless causal masking with a finite -1e30 sentinel.
__global__ __launch_bounds__(128) void attention_kernel(
    const __bf16* __restrict__ q, const __bf16* __restrict__ k,
    const __bf16* __restrict__ vT, __bf16* __restrict__ ctx,
    float* __restrict__ att) {
  __shared__ __attribute__((aligned(16))) __bf16 pbuf[4][16][40];
  int tid = threadIdx.x;
  int lane = tid & 31;
  int hi = (lane >> 4) & 1, lo = lane & 15;
  int wave = __builtin_amdgcn_readfirstlane(tid >> 5);
  int flat = blockIdx.x * 4 + wave;            // scalar
  int b = flat >> 11;
  int rem = flat & 2047;
  int h = rem >> 7;
  int qt = rem & 127;                          // scalar loop bound
  int qbase = qt * 16;
  size_t bh = (size_t)(b * NH + h);

  const __bf16* qrow = q + (bh * SEQ + qbase + lo) * HD;   // Q pre-scaled by 1/8
  v16bf qa0 = load_a16(qrow, hi);              // hd 0..31
  v16bf qa1 = load_a16(qrow + 32, hi);         // hd 32..63
  const __bf16* kbp = k + bh * SEQ * HD;

  float mrow[8], lrow[8];
#pragma unroll
  for (int j = 0; j < 8; ++j) { mrow[j] = NEGBIG; lrow[j] = 0.f; }

  // ---- pass 1: online (m,l), next-tile K prefetch
  v16bf kb0 = load_b16(kbp + (size_t)lo * HD, hi);
  v16bf kb1 = load_b16(kbp + (size_t)lo * HD + 32, hi);
  for (int kt = 0; kt <= qt; ++kt) {
    v16bf nb0 = kb0, nb1 = kb1;
    if (kt < qt) {
      const __bf16* nr = kbp + (size_t)((kt + 1) * 16 + lo) * HD;
      nb0 = load_b16(nr, hi);
      nb1 = load_b16(nr + 32, hi);
    }
    v8f s = {};
    s = wmma_bf16(qa0, kb0, s);
    s = wmma_bf16(qa1, kb1, s);
    int key = kt * 16 + lo;
#pragma unroll
    for (int j = 0; j < 8; ++j) {
      int query = qbase + j + 8 * hi;
      float sc = (key <= query) ? s[j] : NEGBIG;   // branchless mask
      float nm = fmaxf(mrow[j], sc);
      lrow[j] = lrow[j] * __expf(mrow[j] - nm) + __expf(sc - nm);
      mrow[j] = nm;
    }
    kb0 = nb0; kb1 = nb1;
  }
  // combine across 16 lanes per half; all-finite math, sentinel lanes get
  // weight exp(-1e30 - m_finite) == 0 so their pollution vanishes.
#pragma unroll
  for (int off = 1; off < 16; off <<= 1) {
#pragma unroll
    for (int j = 0; j < 8; ++j) {
      float om = __shfl_xor(mrow[j], off, 32);
      float ol = __shfl_xor(lrow[j], off, 32);
      float nm = fmaxf(mrow[j], om);
      lrow[j] = lrow[j] * __expf(mrow[j] - nm) + ol * __expf(om - nm);
      mrow[j] = nm;
    }
  }
  float invl[8];
#pragma unroll
  for (int j = 0; j < 8; ++j) invl[j] = 1.0f / lrow[j];

  // ---- pass 2: probabilities out + ctx (all fragment loads hoisted per pair)
  v8f cacc[4] = {};
  for (int kp = 0; kp <= qt; kp += 2) {
    bool two = (kp + 1 <= qt);                 // scalar -> uniform branch
    v16bf vb[4];
#pragma unroll
    for (int d = 0; d < 4; ++d)
      vb[d] = load_b16(vT + (bh * HD + d * 16 + lo) * SEQ + kp * 16, hi);
    const __bf16* kr0 = kbp + (size_t)(kp * 16 + lo) * HD;
    v16bf kb00 = load_b16(kr0, hi);
    v16bf kb01 = load_b16(kr0 + 32, hi);
    v16bf kb10 = {}, kb11 = {};
    if (two) {
      const __bf16* kr1 = kbp + (size_t)((kp + 1) * 16 + lo) * HD;
      kb10 = load_b16(kr1, hi);
      kb11 = load_b16(kr1 + 32, hi);
    }
    {
      v8f s = {};
      s = wmma_bf16(qa0, kb00, s);
      s = wmma_bf16(qa1, kb01, s);
      int key = kp * 16 + lo;
#pragma unroll
      for (int j = 0; j < 8; ++j) {
        int query = qbase + j + 8 * hi;
        float sc = (key <= query) ? s[j] : NEGBIG;
        float p = __expf(sc - mrow[j]) * invl[j];
        att[(bh * SEQ + query) * SEQ + key] = p;
        pbuf[wave][j + 8 * hi][lo] = (__bf16)p;
      }
    }
    if (two) {
      v8f s = {};
      s = wmma_bf16(qa0, kb10, s);
      s = wmma_bf16(qa1, kb11, s);
      int key = (kp + 1) * 16 + lo;
#pragma unroll
      for (int j = 0; j < 8; ++j) {
        int query = qbase + j + 8 * hi;
        float sc = (key <= query) ? s[j] : NEGBIG;
        float p = __expf(sc - mrow[j]) * invl[j];
        att[(bh * SEQ + query) * SEQ + key] = p;
        pbuf[wave][j + 8 * hi][16 + lo] = (__bf16)p;
      }
    } else {
#pragma unroll
      for (int j = 0; j < 8; ++j)
        pbuf[wave][j + 8 * hi][16 + lo] = (__bf16)0.0f;
    }
    // same-wave cross-lane LDS RAW: wait DS stores before fragment reads
    asm volatile("s_wait_dscnt 0" ::: "memory");
    v16bf pf = load_a16(&pbuf[wave][lo][0], hi);
#pragma unroll
    for (int d = 0; d < 4; ++d) cacc[d] = wmma_bf16(pf, vb[d], cacc[d]);
  }

#pragma unroll
  for (int d = 0; d < 4; ++d)
#pragma unroll
    for (int j = 0; j < 8; ++j) {
      int row = qbase + j + 8 * hi;
      int col = h * HD + d * 16 + lo;
      ctx[((size_t)b * SEQ + row) * DM + col] = (__bf16)cacc[d][j];
    }
}

// ---------------------------------------------------------------- layernorm
template <bool WRITE_B>
__global__ __launch_bounds__(256) void layernorm_kernel(
    const float* __restrict__ in, const float* __restrict__ g,
    const float* __restrict__ bt, float* __restrict__ outF,
    __bf16* __restrict__ outB) {
  __shared__ float s1[256], s2[256];
  int row = blockIdx.x, t = threadIdx.x;
  float v[4], a = 0.f, b2 = 0.f;
#pragma unroll
  for (int i = 0; i < 4; ++i) {
    v[i] = in[(size_t)row * DM + t + i * 256];
    a += v[i]; b2 += v[i] * v[i];
  }
  s1[t] = a; s2[t] = b2;
  __syncthreads();
  for (int off = 128; off > 0; off >>= 1) {
    if (t < off) { s1[t] += s1[t + off]; s2[t] += s2[t + off]; }
    __syncthreads();
  }
  float mean = s1[0] * (1.0f / DM);
  float var  = s2[0] * (1.0f / DM) - mean * mean;
  float rstd = rsqrtf(var + LNEPS);
#pragma unroll
  for (int i = 0; i < 4; ++i) {
    int c = t + i * 256;
    float y = (v[i] - mean) * rstd * g[c] + bt[c];
    outF[(size_t)row * DM + c] = y;
    if (WRITE_B) outB[(size_t)row * DM + c] = (__bf16)y;
  }
}

// ---------------------------------------------------------------- launch
extern "C" void kernel_launch(void* const* d_in, const int* in_sizes, int n_in,
                              void* d_out, int out_size, void* d_ws, size_t ws_size,
                              hipStream_t stream) {
  (void)in_sizes; (void)n_in; (void)out_size; (void)ws_size;
  const float* x   = (const float*)d_in[0];
  const float* wq  = (const float*)d_in[1];
  const float* wk  = (const float*)d_in[2];
  const float* wv  = (const float*)d_in[3];
  const float* wo  = (const float*)d_in[4];
  const float* w1  = (const float*)d_in[5];
  const float* b1  = (const float*)d_in[6];
  const float* w2  = (const float*)d_in[7];
  const float* b2  = (const float*)d_in[8];
  const float* g1  = (const float*)d_in[9];
  const float* bt1 = (const float*)d_in[10];
  const float* g2  = (const float*)d_in[11];
  const float* bt2 = (const float*)d_in[12];

  char* ws = (char*)d_ws;
  auto alloc = [&](size_t bytes) {
    char* p = ws;
    ws += (bytes + 255) & ~(size_t)255;
    return p;
  };
  __bf16* xb  = (__bf16*)alloc((size_t)TOK * DM * 2);
  __bf16* wqT = (__bf16*)alloc((size_t)DM * DM * 2);
  __bf16* wkT = (__bf16*)alloc((size_t)DM * DM * 2);
  __bf16* wvT = (__bf16*)alloc((size_t)DM * DM * 2);
  __bf16* woT = (__bf16*)alloc((size_t)DM * DM * 2);
  __bf16* w1T = (__bf16*)alloc((size_t)DM * DM * 2);
  __bf16* w2T = (__bf16*)alloc((size_t)DM * DM * 2);
  __bf16* qb  = (__bf16*)alloc((size_t)TOK * DM * 2);
  __bf16* kb  = (__bf16*)alloc((size_t)TOK * DM * 2);
  __bf16* vTb = (__bf16*)alloc((size_t)TOK * DM * 2);
  __bf16* ctx = (__bf16*)alloc((size_t)TOK * DM * 2);
  float*  tmpF = (float*)alloc((size_t)TOK * DM * 4);
  float*  x1f  = (float*)alloc((size_t)TOK * DM * 4);
  __bf16* x1b  = (__bf16*)alloc((size_t)TOK * DM * 2);
  __bf16* h1b  = (__bf16*)alloc((size_t)TOK * DM * 2);

  float* outMain = (float*)d_out;
  float* attOut  = (float*)d_out + OUT0_ELEMS;

  dim3 gGemm(TOK / 128, DM / 128);
  dim3 gTr(DM / 32, DM / 32), bTr(32, 8);

  convert_f32_bf16<<<(TOK * DM / 4) / 256, 256, 0, stream>>>(x, xb);
  transpose_w<<<gTr, bTr, 0, stream>>>(wq, wqT);
  transpose_w<<<gTr, bTr, 0, stream>>>(wk, wkT);
  transpose_w<<<gTr, bTr, 0, stream>>>(wv, wvT);
  transpose_w<<<gTr, bTr, 0, stream>>>(wo, woT);
  transpose_w<<<gTr, bTr, 0, stream>>>(w1, w1T);
  transpose_w<<<gTr, bTr, 0, stream>>>(w2, w2T);

  gemm_bf16<0><<<gGemm, 256, 0, stream>>>(xb, wqT, nullptr, nullptr, nullptr, qb,
                                          0.125f);
  gemm_bf16<0><<<gGemm, 256, 0, stream>>>(xb, wkT, nullptr, nullptr, nullptr, kb,
                                          1.0f);
  gemm_bf16<1><<<gGemm, 256, 0, stream>>>(xb, wvT, nullptr, nullptr, nullptr, vTb,
                                          1.0f);

  zero_f4<<<(unsigned)((ATT_ELEMS / 16) / 256), 256, 0, stream>>>((float4*)attOut);
  attention_kernel<<<(BATCH * NH * (SEQ / 16)) / 4, 128, 0, stream>>>(qb, kb, vTb, ctx,
                                                                     attOut);

  gemm_bf16<2><<<gGemm, 256, 0, stream>>>(ctx, woT, nullptr, x, tmpF, nullptr, 1.0f);
  layernorm_kernel<true><<<TOK, 256, 0, stream>>>(tmpF, g1, bt1, x1f, x1b);
  gemm_bf16<3><<<gGemm, 256, 0, stream>>>(x1b, w1T, b1, nullptr, nullptr, h1b, 1.0f);
  gemm_bf16<4><<<gGemm, 256, 0, stream>>>(h1b, w2T, b2, x1f, tmpF, nullptr, 1.0f);
  layernorm_kernel<false><<<TOK, 256, 0, stream>>>(tmpF, g2, bt2, outMain, nullptr);
}
